// DualGNNInterface_1417339208210
// MI455X (gfx1250) — matrix operand order repaired and checked by reading, not compiled, not run on hardware
//
#include <hip/hip_runtime.h>
#include <hip/hip_bf16.h>

typedef __attribute__((ext_vector_type(16))) __bf16 v16bf;
typedef __attribute__((ext_vector_type(8)))  __bf16 v8bf;
typedef __attribute__((ext_vector_type(4)))  __bf16 v4bf;
typedef __attribute__((ext_vector_type(8)))  float  v8f;

#define GEMM_BIAS    1
#define GEMM_ACCUM   2
#define GEMM_BF16OUT 4

// ---------------------------------------------------------------------------
// Weight renormalization: w_eff = w / (EPS + norm/sqrt(128)) * (1/sqrt(128))
// 16 matrices of [128,128] packed into one bf16 buffer:
//   0-1 mp_in, 2-7 sage_l (layer-major), 8-13 sage_r, 14-15 mp_out
// ---------------------------------------------------------------------------
__global__ __launch_bounds__(128) void prep_weights(
    const float* __restrict__ mp_in, const float* __restrict__ lw,
    const float* __restrict__ rw, const float* __restrict__ mp_out,
    __bf16* __restrict__ W16)
{
  int mat = blockIdx.x >> 7;
  int row = blockIdx.x & 127;
  int c   = threadIdx.x;
  const float* src;
  if      (mat < 2)  src = mp_in  + (size_t)mat * 16384;
  else if (mat < 8)  src = lw     + (size_t)(mat - 2) * 16384;
  else if (mat < 14) src = rw     + (size_t)(mat - 8) * 16384;
  else               src = mp_out + (size_t)(mat - 14) * 16384;

  float w = src[row * 128 + c];
  __shared__ float red[128];
  red[c] = w * w;
  __syncthreads();
  #pragma unroll
  for (int s = 64; s > 0; s >>= 1) {
    if (c < s) red[c] += red[c + s];
    __syncthreads();
  }
  float norm = sqrtf(red[0]);
  const float inv_sqrtC = 0.08838834764831845f; // 1/sqrt(128)
  float scale = inv_sqrtC / (1e-4f + inv_sqrtC * norm);
  W16[(size_t)mat * 16384 + row * 128 + c] = (__bf16)(w * scale);
}

// [B,C,H,W] f32 -> node-major [B*H*W, C] bf16  (B=nimg/16384, C=H=W=128)
__global__ __launch_bounds__(256) void img_to_nodes(
    const float* __restrict__ x, __bf16* __restrict__ ximg, int nimg)
{
  int idx = blockIdx.x * 256 + threadIdx.x;
  if (idx >= nimg * 128) return;
  int c = idx / nimg;            // channel slow -> reads coalesced over w
  int m = idx - c * nimg;
  int b = m >> 14, h = (m >> 7) & 127, w = m & 127;
  size_t in = ((size_t)((b * 128 + c) * 128 + h) << 7) | w;
  ximg[(size_t)m * 128 + c] = (__bf16)x[in];
}

__global__ __launch_bounds__(256) void f32_to_bf16_kernel(
    const float* __restrict__ in, __bf16* __restrict__ out, int n)
{
  int i = blockIdx.x * 256 + threadIdx.x;
  if (i < n) out[i] = (__bf16)in[i];
}

// ---------------------------------------------------------------------------
// BF16 WMMA GEMM:  Y[M,128] = X[M,128] @ W^T   (W row-major [N=128, K=128])
// 256 threads = 8 waves; wave w owns rows [m0, m0+16); weight staged in LDS.
// ---------------------------------------------------------------------------
__global__ __launch_bounds__(256) void gemm128_bf16(
    const __bf16* __restrict__ X, const __bf16* __restrict__ W,
    const float* __restrict__ bias,
    float* __restrict__ Yf, __bf16* __restrict__ Yb,
    int M, int flags)
{
  __shared__ __bf16 lw[128 * 128];            // 32 KB of 320 KB WGP LDS
  int tid = threadIdx.x;
  for (int i = tid; i < 2048; i += 256)        // cooperative 16B-chunk fill
    ((float4*)lw)[i] = ((const float4*)W)[i];
  __syncthreads();

  int wave = tid >> 5;
  int lane = tid & 31;
  int hi   = lane >> 4;                        // 0: lanes 0-15, 1: lanes 16-31
  int m0   = blockIdx.x * 128 + wave * 16;
  if (m0 >= M) return;
  int mrow = m0 + (lane & 15);
  const __bf16* xrow = X + (size_t)mrow * 128;

  v8f acc[8] = {};

  #pragma unroll
  for (int kt = 0; kt < 4; ++kt) {
    // A tile (16x32 bf16): lane M = lane%16; 16 contiguous K per lane half
    int kbaseA = kt * 32 + hi * 16;
    v8bf alo = *(const v8bf*)(xrow + kbaseA);
    v8bf ahi = *(const v8bf*)(xrow + kbaseA + 8);
    v16bf a = __builtin_shufflevector(alo, ahi,
        0,1,2,3,4,5,6,7, 8,9,10,11,12,13,14,15);
    int kbaseB = kt * 32 + hi * 16;
    #pragma unroll
    for (int nt = 0; nt < 8; ++nt) {
      // B tile (32x16): lane N = lane%16; 16 contiguous K per lane half
      int n = nt * 16 + (lane & 15);
      v16bf b = *(const v16bf*)(lw + n * 128 + kbaseB);
      acc[nt] = __builtin_amdgcn_wmma_f32_16x16x32_bf16(
          false, a, false, b, (short)0, acc[nt], false, false);
    }
  }

  // D layout: VGPR r -> row m0 + r + 8*hi; lane%16 -> column within N-tile
  #pragma unroll
  for (int nt = 0; nt < 8; ++nt) {
    int col = nt * 16 + (lane & 15);
    float bv = (flags & GEMM_BIAS) ? bias[col] : 0.0f;
    #pragma unroll
    for (int r = 0; r < 8; ++r) {
      int row = m0 + r + hi * 8;
      size_t o = (size_t)row * 128 + col;
      float v = acc[nt][r] + bv;
      if (flags & GEMM_ACCUM)   v += Yf[o];
      if (flags & GEMM_BF16OUT) Yb[o] = (__bf16)v;
      else                      Yf[o] = v;
    }
  }
}

// One wave per edge; lane handles 4 channels; hardware f32 atomics.
__global__ __launch_bounds__(256) void scatter_mean(
    const __bf16* __restrict__ xsrc, const int* __restrict__ src,
    const int* __restrict__ dst, float* __restrict__ sum,
    float* __restrict__ cnt, int E)
{
  int t = blockIdx.x * 256 + threadIdx.x;
  int e = t >> 5, lane = t & 31;
  if (e >= E) return;
  int s = src[e], d = dst[e];
  v4bf v = *(const v4bf*)(xsrc + (size_t)s * 128 + lane * 4);
  float* q = sum + (size_t)d * 128 + lane * 4;
  unsafeAtomicAdd(q + 0, (float)v[0]);
  unsafeAtomicAdd(q + 1, (float)v[1]);
  unsafeAtomicAdd(q + 2, (float)v[2]);
  unsafeAtomicAdd(q + 3, (float)v[3]);
  if (lane == 0) unsafeAtomicAdd(cnt + d, 1.0f);
}

__global__ __launch_bounds__(256) void finalize_agg(
    const float* __restrict__ sum, const float* __restrict__ cnt,
    __bf16* __restrict__ agg, int M)
{
  int i = blockIdx.x * 256 + threadIdx.x;     // one thread per 4 channels
  if (i >= M * 32) return;
  int m = i >> 5, c = (i & 31) * 4;
  float inv = 1.0f / fmaxf(cnt[m], 1.0f);
  const float* s = sum + (size_t)m * 128 + c;
  v4bf o;
  o[0] = (__bf16)(s[0] * inv); o[1] = (__bf16)(s[1] * inv);
  o[2] = (__bf16)(s[2] * inv); o[3] = (__bf16)(s[3] * inv);
  *(v4bf*)(agg + (size_t)m * 128 + c) = o;
}

// Row L2-normalize O1 (and optionally O2), hetero-sum, emit bf16 features.
__global__ __launch_bounds__(128) void row_norm_combine(
    const float* __restrict__ O1, const float* __restrict__ O2,
    __bf16* __restrict__ xout, int M)
{
  int m = blockIdx.x, c = threadIdx.x;
  __shared__ float red[128];
  float v1 = O1[(size_t)m * 128 + c];
  red[c] = v1 * v1;
  __syncthreads();
  #pragma unroll
  for (int s = 64; s > 0; s >>= 1) { if (c < s) red[c] += red[c + s]; __syncthreads(); }
  float n1 = fmaxf(sqrtf(red[0]), 1e-12f);
  __syncthreads();
  float out = v1 / n1;
  if (O2) {
    float v2 = O2[(size_t)m * 128 + c];
    red[c] = v2 * v2;
    __syncthreads();
    #pragma unroll
    for (int s = 64; s > 0; s >>= 1) { if (c < s) red[c] += red[c + s]; __syncthreads(); }
    float n2 = fmaxf(sqrtf(red[0]), 1e-12f);
    out += v2 / n2;
  }
  xout[(size_t)m * 128 + c] = (__bf16)out;
}

// node-major [B*H*W,128] f32 -> [B,128,128,128] f32
__global__ __launch_bounds__(256) void nodes_to_nchw(
    const float* __restrict__ in, float* __restrict__ out, int total)
{
  int o = blockIdx.x * 256 + threadIdx.x;
  if (o >= total) return;
  int w = o & 127, h = (o >> 7) & 127, c = (o >> 14) & 127, b = o >> 21;
  size_t m = ((size_t)((b * 128 + h) * 128 + w));
  out[o] = in[m * 128 + c];
}

extern "C" void kernel_launch(void* const* d_in, const int* in_sizes, int n_in,
                              void* d_out, int out_size, void* d_ws, size_t ws_size,
                              hipStream_t stream) {
  const float* x        = (const float*)d_in[0];
  const float* x_class  = (const float*)d_in[1];
  const int*   src_ii   = (const int*)d_in[2];
  const int*   dst_ii   = (const int*)d_in[3];
  const int*   src_ci   = (const int*)d_in[4];
  const int*   dst_ci   = (const int*)d_in[5];
  const int*   src_ic   = (const int*)d_in[6];
  const int*   dst_ic   = (const int*)d_in[7];
  const float* mp_in_w  = (const float*)d_in[8];
  const float* sage_l_w = (const float*)d_in[9];
  const float* sage_l_b = (const float*)d_in[10];
  const float* sage_r_w = (const float*)d_in[11];
  const float* mp_out_w = (const float*)d_in[12];

  const int NIMG = in_sizes[0] / 128;
  const int NCLS = in_sizes[1] / 128;
  const int E_ii = in_sizes[2], E_ci = in_sizes[4], E_ic = in_sizes[6];

  char* ws = (char*)d_ws;
  size_t off = 0;
  auto alloc = [&](size_t bytes) -> char* {
    char* p = ws + off;
    off = (off + bytes + 255) & ~(size_t)255;
    return p;
  };
  __bf16* W16     = (__bf16*)alloc((size_t)16 * 16384 * 2);
  __bf16* ximg_bf = (__bf16*)alloc((size_t)NIMG * 128 * 2);
  __bf16* xcls_bf = (__bf16*)alloc((size_t)NCLS * 128 * 2);
  float*  sumA    = (float*)alloc((size_t)NIMG * 128 * 4);  // reused as O1
  float*  sumB    = (float*)alloc((size_t)NIMG * 128 * 4);  // reused as O2
  float*  sumC    = (float*)alloc((size_t)NCLS * 128 * 4);  // reused as O3
  float*  cntA    = (float*)alloc((size_t)NIMG * 4);
  float*  cntB    = (float*)alloc((size_t)NIMG * 4);
  float*  cntC    = (float*)alloc((size_t)NCLS * 4);
  __bf16* aggA    = (__bf16*)alloc((size_t)NIMG * 128 * 2);
  __bf16* aggB    = (__bf16*)alloc((size_t)NIMG * 128 * 2);
  __bf16* aggC    = (__bf16*)alloc((size_t)NCLS * 128 * 2);

  // 1) weights
  prep_weights<<<2048, 128, 0, stream>>>(mp_in_w, sage_l_w, sage_r_w, mp_out_w, W16);
  // 2) features to node-major bf16
  img_to_nodes<<<(NIMG * 128 + 255) / 256, 256, 0, stream>>>(x, ximg_bf, NIMG);
  f32_to_bf16_kernel<<<(NCLS * 128 + 255) / 256, 256, 0, stream>>>(x_class, xcls_bf, NCLS * 128);
  // 3) input MPConv (in-place is safe: row m depends only on row m)
  gemm128_bf16<<<NIMG / 128, 256, 0, stream>>>(ximg_bf, W16 + 0 * 16384, nullptr,
                                               nullptr, ximg_bf, NIMG, GEMM_BF16OUT);
  gemm128_bf16<<<NCLS / 128, 256, 0, stream>>>(xcls_bf, W16 + 1 * 16384, nullptr,
                                               nullptr, xcls_bf, NCLS, GEMM_BF16OUT);

  for (int layer = 0; layer < 2; ++layer) {
    hipMemsetAsync(sumA, 0, (size_t)NIMG * 128 * 4, stream);
    hipMemsetAsync(sumB, 0, (size_t)NIMG * 128 * 4, stream);
    hipMemsetAsync(sumC, 0, (size_t)NCLS * 128 * 4, stream);
    hipMemsetAsync(cntA, 0, (size_t)NIMG * 4, stream);
    hipMemsetAsync(cntB, 0, (size_t)NIMG * 4, stream);
    hipMemsetAsync(cntC, 0, (size_t)NCLS * 4, stream);

    scatter_mean<<<(E_ii * 32 + 255) / 256, 256, 0, stream>>>(ximg_bf, src_ii, dst_ii, sumA, cntA, E_ii);
    scatter_mean<<<(E_ci * 32 + 255) / 256, 256, 0, stream>>>(xcls_bf, src_ci, dst_ci, sumB, cntB, E_ci);
    scatter_mean<<<(E_ic * 32 + 255) / 256, 256, 0, stream>>>(ximg_bf, src_ic, dst_ic, sumC, cntC, E_ic);

    finalize_agg<<<(NIMG * 32 + 255) / 256, 256, 0, stream>>>(sumA, cntA, aggA, NIMG);
    finalize_agg<<<(NIMG * 32 + 255) / 256, 256, 0, stream>>>(sumB, cntB, aggB, NIMG);
    finalize_agg<<<(NCLS * 32 + 255) / 256, 256, 0, stream>>>(sumC, cntC, aggC, NCLS);

    const __bf16* lw0 = W16 + (size_t)(2 + layer * 3 + 0) * 16384;
    const __bf16* lw1 = W16 + (size_t)(2 + layer * 3 + 1) * 16384;
    const __bf16* lw2 = W16 + (size_t)(2 + layer * 3 + 2) * 16384;
    const __bf16* rw0 = W16 + (size_t)(8 + layer * 3 + 0) * 16384;
    const __bf16* rw1 = W16 + (size_t)(8 + layer * 3 + 1) * 16384;
    const __bf16* rw2 = W16 + (size_t)(8 + layer * 3 + 2) * 16384;
    const float* b0 = sage_l_b + (size_t)(layer * 3 + 0) * 128;
    const float* b1 = sage_l_b + (size_t)(layer * 3 + 1) * 128;
    const float* b2 = sage_l_b + (size_t)(layer * 3 + 2) * 128;

    // o = agg @ lw^T + b + x_dst @ rw^T   (sum buffers now free -> reuse as O)
    gemm128_bf16<<<NIMG / 128, 256, 0, stream>>>(aggA, lw0, b0, sumA, nullptr, NIMG, GEMM_BIAS);
    gemm128_bf16<<<NIMG / 128, 256, 0, stream>>>(ximg_bf, rw0, nullptr, sumA, nullptr, NIMG, GEMM_ACCUM);
    gemm128_bf16<<<NIMG / 128, 256, 0, stream>>>(aggB, lw1, b1, sumB, nullptr, NIMG, GEMM_BIAS);
    gemm128_bf16<<<NIMG / 128, 256, 0, stream>>>(ximg_bf, rw1, nullptr, sumB, nullptr, NIMG, GEMM_ACCUM);
    gemm128_bf16<<<NCLS / 128, 256, 0, stream>>>(aggC, lw2, b2, sumC, nullptr, NCLS, GEMM_BIAS);
    gemm128_bf16<<<NCLS / 128, 256, 0, stream>>>(xcls_bf, rw2, nullptr, sumC, nullptr, NCLS, GEMM_ACCUM);

    // L2-normalize rows; x_img' = n(o_ii) + n(o_ci); x_cls' = n(o_ic)
    row_norm_combine<<<NIMG, 128, 0, stream>>>(sumA, sumB, ximg_bf, NIMG);
    row_norm_combine<<<NCLS, 128, 0, stream>>>(sumC, nullptr, xcls_bf, NCLS);
  }

  // output MPConv
  gemm128_bf16<<<NIMG / 128, 256, 0, stream>>>(ximg_bf, W16 + 14 * 16384, nullptr,
                                               sumA, nullptr, NIMG, 0);
  nodes_to_nchw<<<(NIMG * 128 + 255) / 256, 256, 0, stream>>>(sumA, (float*)d_out, NIMG * 128);
  gemm128_bf16<<<NCLS / 128, 256, 0, stream>>>(xcls_bf, W16 + 15 * 16384, nullptr,
                                               (float*)d_out + (size_t)NIMG * 128, nullptr, NCLS, 0);
}
